// EncoderLayer_28381143892639
// MI455X (gfx1250) — matrix operand order repaired and checked by reading, not compiled, not run on hardware
//
#include <hip/hip_runtime.h>
#include <hip/hip_bf16.h>
#include <cstdint>
#include <cstddef>

// Problem constants (match reference)
#define BB   8
#define SS   1024
#define DDIM 512
#define DFF  2048
#define HH   8
#define DHH  64

typedef __attribute__((ext_vector_type(16))) __bf16 v16bf;
typedef __attribute__((ext_vector_type(8)))  float  v8f;

union BF16Frag { v16bf v; unsigned short u[16]; uint4 q[2]; };

__device__ __forceinline__ unsigned short f32_to_bf16(float f) {
  union { __hip_bfloat16 b; unsigned short u; } cv;
  cv.b = __float2bfloat16(f);   // single v_cvt op, RNE
  return cv.u;
}

// CDNA5 async global->LDS copy (16B per lane), tracked by ASYNCcnt.
// Generic flat address of a __shared__ object has the LDS byte offset in its
// low 32 bits (ISA 10.2: LDS aperture truncates addr[31:0]).
__device__ __forceinline__ void async_copy_b128(void* lds_ptr, const void* gptr) {
  uint32_t lo = (uint32_t)(uintptr_t)lds_ptr;
  asm volatile("global_load_async_to_lds_b128 %0, %1, off"
               : : "v"(lo), "v"(gptr) : "memory");
}
__device__ __forceinline__ void wait_async0() {
  asm volatile("s_wait_asynccnt 0" ::: "memory");
}

// --------------------------------------------------------------------------
// Generic bf16 GEMM with TRANSPOSED weights, async double-buffered k-loop:
//   C[M,N] = A[M,K] * W[K,N],  B operand passed as Wt[N,K] (row-major N x K)
// EPI: 0 = store f32; 1 = store bf16; 2 = +bias, ReLU, store bf16;
//      3 = +bias, store f32; 4 = *0.125, store bf16 (pre-scaled Q)
// Tile: 128x128x32, 128 threads (4 waves), wave -> 64x64 (4x4 wmma tiles).
// Next tile streams HBM->LDS via async copies while WMMAs run on current.
// Requires M%128==0, N%128==0, K%32==0.
// --------------------------------------------------------------------------
template <int EPI>
__global__ __launch_bounds__(128) void gemm_bf16_kernel(
    const unsigned short* __restrict__ A, const unsigned short* __restrict__ Bt,
    void* __restrict__ C, const float* __restrict__ bias, int M, int N, int K) {
  __shared__ alignas(16) unsigned short As[2][128][40];   // [buf][m][k]
  __shared__ alignas(16) unsigned short Bs[2][128][40];   // [buf][n][k]

  const int tid  = threadIdx.x;
  const int lane = tid & 31;
  const int wave = tid >> 5;
  const int bm   = blockIdx.y * 128;
  const int bn   = blockIdx.x * 128;
  const int wm   = (wave >> 1) * 64;
  const int wn   = (wave & 1) * 64;
  const int lo   = lane & 15;
  const int hi8  = (lane >> 4) << 3;   // 0 or 8
  const int hi16 = (lane >> 4) << 4;   // 0 or 16

  const unsigned short* rowA = A  + (size_t)(bm + tid) * K;
  const unsigned short* rowB = Bt + (size_t)(bn + tid) * K;

  v8f acc[4][4];
#pragma unroll
  for (int i = 0; i < 4; ++i)
#pragma unroll
    for (int j = 0; j < 4; ++j) acc[i][j] = (v8f){0.f,0.f,0.f,0.f,0.f,0.f,0.f,0.f};

  const int nk = K >> 5;
  {   // async preload of tile 0 (4 x 16B per thread per operand)
#pragma unroll
    for (int i = 0; i < 4; ++i) {
      async_copy_b128(&As[0][tid][i * 8], rowA + i * 8);
      async_copy_b128(&Bs[0][tid][i * 8], rowB + i * 8);
    }
  }
  wait_async0();
  __syncthreads();

  for (int it = 0; it < nk; ++it) {
    const int cur = it & 1, nxt = cur ^ 1;
    if (it + 1 < nk) {   // stream next tile into alternate buffer (no VGPRs)
      const unsigned short* ga = rowA + (it + 1) * 32;
      const unsigned short* gb = rowB + (it + 1) * 32;
#pragma unroll
      for (int i = 0; i < 4; ++i) {
        async_copy_b128(&As[nxt][tid][i * 8], ga + i * 8);
        async_copy_b128(&Bs[nxt][tid][i * 8], gb + i * 8);
      }
    }

    BF16Frag af[4];
#pragma unroll
    for (int ti = 0; ti < 4; ++ti) {
      int m = wm + ti * 16 + lo;
      af[ti].q[0] = *(const uint4*)&As[cur][m][hi8];        // k = hi8 .. hi8+7
      af[ti].q[1] = *(const uint4*)&As[cur][m][16 + hi8];   // k = 16+hi8 ..
    }
#pragma unroll
    for (int tj = 0; tj < 4; ++tj) {
      BF16Frag bf;
      int n = wn + tj * 16 + lo;
      bf.q[0] = *(const uint4*)&Bs[cur][n][hi8];            // k = hi8 .. hi8+7
      bf.q[1] = *(const uint4*)&Bs[cur][n][16 + hi8];       // k = 16+hi8 ..
#pragma unroll
      for (int ti = 0; ti < 4; ++ti)
        acc[ti][tj] = __builtin_amdgcn_wmma_f32_16x16x32_bf16(
            false, af[ti].v, false, bf.v, (short)0, acc[ti][tj], false, false);
    }

    wait_async0();
    __syncthreads();
  }

#pragma unroll
  for (int ti = 0; ti < 4; ++ti)
#pragma unroll
    for (int tj = 0; tj < 4; ++tj) {
      int n = bn + wn + tj * 16 + lo;
#pragma unroll
      for (int v = 0; v < 8; ++v) {
        int m = bm + wm + ti * 16 + v + hi8;
        float val = acc[ti][tj][v];
        if (EPI == 2 || EPI == 3) val += bias[n];
        if (EPI == 2) val = val > 0.f ? val : 0.f;
        if (EPI == 4) val *= 0.125f;
        if (EPI == 0 || EPI == 3)
          ((float*)C)[(size_t)m * N + n] = val;
        else
          ((unsigned short*)C)[(size_t)m * N + n] = f32_to_bf16(val);
      }
    }
}

// --------------------------------------------------------------------------
// Flash attention: per (q-tile of 64, head, batch). Q,K,V bf16 [B*S, D] with
// head slice at column h*DH. Q is pre-scaled by 1/sqrt(DH) in its GEMM.
// K double-buffered via async copies; V transposed through short-lived regs
// into the alternate LDS buffer so P*V fragments are contiguous.
// --------------------------------------------------------------------------
__global__ __launch_bounds__(128) void attn_kernel(
    const unsigned short* __restrict__ Q, const unsigned short* __restrict__ K,
    const unsigned short* __restrict__ V, unsigned short* __restrict__ O) {
  __shared__ alignas(16) unsigned short Qs[64][72];          // [q][d]
  __shared__ alignas(16) unsigned short Ks[2][32][72];       // [buf][kpos][d]
  __shared__ alignas(16) unsigned short Vt[2][64][40];       // [buf][d][kpos]
  __shared__ alignas(16) unsigned short Ps[4][16][40];       // per-wave P scratch

  const int tid  = threadIdx.x;
  const int lane = tid & 31;
  const int wave = tid >> 5;
  const int q0   = blockIdx.x * 64;
  const int h    = blockIdx.y;
  const int b    = blockIdx.z;
  const int lo   = lane & 15;
  const int hi8  = (lane >> 4) << 3;
  const int hi16 = (lane >> 4) << 4;
  const size_t colbase = (size_t)h * DHH;
  const int sr = tid >> 2, sc = (tid & 3) * 16;   // staging row/col for K,V

  {   // async K block 0; Q tile + V block 0 via regular path
    const unsigned short* kg = K + ((size_t)b * SS + sr) * DDIM + colbase + sc;
    async_copy_b128(&Ks[0][sr][sc], kg);
    async_copy_b128(&Ks[0][sr][sc + 8], kg + 8);

    int r = tid >> 1, c = (tid & 1) * 32;
    const uint4* src = (const uint4*)(Q + ((size_t)b * SS + q0 + r) * DDIM + colbase + c);
    uint4* dst = (uint4*)&Qs[r][c];
#pragma unroll
    for (int i = 0; i < 4; ++i) dst[i] = src[i];

    BF16Frag vtmp;
    const uint4* vsrc = (const uint4*)(V + ((size_t)b * SS + sr) * DDIM + colbase + sc);
    vtmp.q[0] = vsrc[0]; vtmp.q[1] = vsrc[1];
#pragma unroll
    for (int i = 0; i < 16; ++i) Vt[0][sc + i][sr] = vtmp.u[i];
  }
  wait_async0();
  __syncthreads();

  float mrow[8], lrow[8];
  v8f o_acc[4];
#pragma unroll
  for (int v = 0; v < 8; ++v) { mrow[v] = -1.0e30f; lrow[v] = 0.f; }
#pragma unroll
  for (int t = 0; t < 4; ++t) o_acc[t] = (v8f){0.f,0.f,0.f,0.f,0.f,0.f,0.f,0.f};

  const int njb = SS / 32;
  for (int jt = 0; jt < njb; ++jt) {
    const int cur = jt & 1, nxt = cur ^ 1;
    if (jt + 1 < njb) {   // stream next K async; transpose next V immediately
      int j = (jt + 1) * 32;
      const unsigned short* kg = K + ((size_t)b * SS + j + sr) * DDIM + colbase + sc;
      async_copy_b128(&Ks[nxt][sr][sc], kg);
      async_copy_b128(&Ks[nxt][sr][sc + 8], kg + 8);
      BF16Frag vtmp;
      const uint4* vsrc = (const uint4*)(V + ((size_t)b * SS + j + sr) * DDIM + colbase + sc);
      vtmp.q[0] = vsrc[0]; vtmp.q[1] = vsrc[1];
#pragma unroll
      for (int i = 0; i < 16; ++i) Vt[nxt][sc + i][sr] = vtmp.u[i];
    }

    // S = Q * K^T over DH=64 (two K=32 chunks), two 16-key tiles
    v8f sfr[2];
    sfr[0] = (v8f){0.f,0.f,0.f,0.f,0.f,0.f,0.f,0.f};
    sfr[1] = sfr[0];
#pragma unroll
    for (int kc = 0; kc < 2; ++kc) {
      BF16Frag qa;
      int m = wave * 16 + lo;
      qa.q[0] = *(const uint4*)&Qs[m][kc * 32 + hi8];
      qa.q[1] = *(const uint4*)&Qs[m][kc * 32 + 16 + hi8];
#pragma unroll
      for (int t = 0; t < 2; ++t) {
        BF16Frag kb;
        int n = t * 16 + lo;   // key position within block
        kb.q[0] = *(const uint4*)&Ks[cur][n][kc * 32 + hi8];
        kb.q[1] = *(const uint4*)&Ks[cur][n][kc * 32 + 16 + hi8];
        sfr[t] = __builtin_amdgcn_wmma_f32_16x16x32_bf16(
            false, qa.v, false, kb.v, (short)0, sfr[t], false, false);
      }
    }

    // online softmax over this 32-key block (Q already scaled)
#pragma unroll
    for (int v = 0; v < 8; ++v) {
      float a0 = sfr[0][v];
      float a1 = sfr[1][v];
      float mx = fmaxf(a0, a1);
#pragma unroll
      for (int off = 1; off < 16; off <<= 1) mx = fmaxf(mx, __shfl_xor(mx, off, 32));
      float mnew  = fmaxf(mrow[v], mx);
      float alpha = __expf(mrow[v] - mnew);
      mrow[v] = mnew;
      float p0 = __expf(a0 - mnew);
      float p1 = __expf(a1 - mnew);
      sfr[0][v] = p0; sfr[1][v] = p1;
      float rs = p0 + p1;
#pragma unroll
      for (int off = 1; off < 16; off <<= 1) rs += __shfl_xor(rs, off, 32);
      lrow[v] = lrow[v] * alpha + rs;
#pragma unroll
      for (int t = 0; t < 4; ++t) o_acc[t][v] *= alpha;
    }

    // relayout P (C-layout f32 -> A-layout bf16) via per-wave LDS scratch
#pragma unroll
    for (int v = 0; v < 8; ++v) {
      Ps[wave][hi8 + v][lo]      = f32_to_bf16(sfr[0][v]);
      Ps[wave][hi8 + v][16 + lo] = f32_to_bf16(sfr[1][v]);
    }
    BF16Frag pa;
    pa.q[0] = *(const uint4*)&Ps[wave][lo][hi8];
    pa.q[1] = *(const uint4*)&Ps[wave][lo][16 + hi8];

    // O += P * V  (4 d-tiles of 16); V fragment contiguous from Vt
#pragma unroll
    for (int t = 0; t < 4; ++t) {
      BF16Frag vb;
      int n = t * 16 + lo;   // d index
      vb.q[0] = *(const uint4*)&Vt[cur][n][hi8];
      vb.q[1] = *(const uint4*)&Vt[cur][n][16 + hi8];
      o_acc[t] = __builtin_amdgcn_wmma_f32_16x16x32_bf16(
          false, pa.v, false, vb.v, (short)0, o_acc[t], false, false);
    }

    wait_async0();
    __syncthreads();
  }

  // normalize and store bf16
#pragma unroll
  for (int v = 0; v < 8; ++v) {
    float inv = 1.0f / lrow[v];
    int m = q0 + wave * 16 + v + hi8;
    size_t row = (size_t)b * SS + m;
#pragma unroll
    for (int t = 0; t < 4; ++t) {
      int d = t * 16 + lo;
      O[row * DDIM + colbase + d] = f32_to_bf16(o_acc[t][v] * inv);
    }
  }
}

// --------------------------------------------------------------------------
// Elementwise helpers
// --------------------------------------------------------------------------
// wt[n*K + k] = bf16( w[k*N + n] )   (convert + transpose, once per launch)
__global__ void cvt_transpose_kernel(const float* __restrict__ w,
                                     unsigned short* __restrict__ wt,
                                     int Kd, int Nd) {
  int t = blockIdx.x * blockDim.x + threadIdx.x;
  if (t >= Kd * Nd) return;
  int n = t / Kd;
  int k = t - n * Kd;
  wt[t] = f32_to_bf16(w[(size_t)k * Nd + n]);
}

// out[b,s,d4*4..] = bf16( x[b, idx[b, flip ? S-1-s : s], d4*4..] ), 4 at a time
__global__ void gather_bf16_kernel(const float* __restrict__ x,
                                   const int* __restrict__ idx,
                                   unsigned short* __restrict__ out, int flip) {
  const int nd4 = DDIM / 4;
  int t = blockIdx.x * blockDim.x + threadIdx.x;
  if (t >= BB * SS * nd4) return;
  int d4 = t % nd4;
  int s  = (t / nd4) % SS;
  int b  = t / (nd4 * SS);
  int ss = flip ? (SS - 1 - s) : s;
  int i  = idx[b * SS + ss];
  float4 v = *(const float4*)&x[((size_t)b * SS + i) * DDIM + d4 * 4];
  union { unsigned short u[4]; uint2 q; } o;
  o.u[0] = f32_to_bf16(v.x); o.u[1] = f32_to_bf16(v.y);
  o.u[2] = f32_to_bf16(v.z); o.u[3] = f32_to_bf16(v.w);
  *(uint2*)&out[(size_t)t * 4] = o.q;
}

// accum[b,s,:] += fwd[b, i, :] + rev[b, S-1-i, :],  i = idx[b,s], 4 at a time
__global__ void addgather_kernel(float* __restrict__ accum,
                                 const float* __restrict__ fwd,
                                 const float* __restrict__ rev,
                                 const int* __restrict__ idx) {
  const int nd4 = DDIM / 4;
  int t = blockIdx.x * blockDim.x + threadIdx.x;
  if (t >= BB * SS * nd4) return;
  int d4 = t % nd4;
  int s  = (t / nd4) % SS;
  int b  = t / (nd4 * SS);
  int i  = idx[b * SS + s];
  float4 a = *(const float4*)&accum[(size_t)t * 4];
  float4 f = *(const float4*)&fwd[((size_t)b * SS + i) * DDIM + d4 * 4];
  float4 r = *(const float4*)&rev[((size_t)b * SS + (SS - 1 - i)) * DDIM + d4 * 4];
  a.x += f.x + r.x; a.y += f.y + r.y; a.z += f.z + r.z; a.w += f.w + r.w;
  *(float4*)&accum[(size_t)t * 4] = a;
}

// LayerNorm over D=512 of (x [+ res]); writes f32 and optional bf16.
__global__ __launch_bounds__(256) void ln_kernel(
    const float* __restrict__ x, const float* __restrict__ res,
    const float* __restrict__ g, const float* __restrict__ beta,
    float* __restrict__ out_f, unsigned short* __restrict__ out_b) {
  __shared__ float red[2][8];
  int row = blockIdx.x;
  const float* xr = x + (size_t)row * DDIM;
  const float* rr = res ? res + (size_t)row * DDIM : nullptr;
  float vals[2], sum = 0.f, sumsq = 0.f;
#pragma unroll
  for (int i = 0; i < 2; ++i) {
    int c = threadIdx.x + i * 256;
    float t = xr[c] + (rr ? rr[c] : 0.f);
    vals[i] = t; sum += t; sumsq += t * t;
  }
#pragma unroll
  for (int off = 1; off < 32; off <<= 1) {
    sum   += __shfl_xor(sum, off, 32);
    sumsq += __shfl_xor(sumsq, off, 32);
  }
  int wv = threadIdx.x >> 5;
  if ((threadIdx.x & 31) == 0) { red[0][wv] = sum; red[1][wv] = sumsq; }
  __syncthreads();
  float ts = 0.f, tq = 0.f;
#pragma unroll
  for (int i = 0; i < 8; ++i) { ts += red[0][i]; tq += red[1][i]; }
  float mean = ts * (1.0f / DDIM);
  float var  = tq * (1.0f / DDIM) - mean * mean;
  float rstd = rsqrtf(var + 1e-5f);
#pragma unroll
  for (int i = 0; i < 2; ++i) {
    int c = threadIdx.x + i * 256;
    float y = (vals[i] - mean) * rstd * g[c] + beta[c];
    if (out_f) out_f[(size_t)row * DDIM + c] = y;
    if (out_b) out_b[(size_t)row * DDIM + c] = f32_to_bf16(y);
  }
}

// --------------------------------------------------------------------------
// Host orchestration
// --------------------------------------------------------------------------
extern "C" void kernel_launch(void* const* d_in, const int* in_sizes, int n_in,
                              void* d_out, int out_size, void* d_ws, size_t ws_size,
                              hipStream_t stream) {
  (void)in_sizes; (void)n_in; (void)out_size; (void)ws_size;

  const float* x        = (const float*)d_in[0];
  const int*   idx_emb[2] = { (const int*)d_in[1], (const int*)d_in[3] };
  const int*   idx_pro[2] = { (const int*)d_in[2], (const int*)d_in[4] };
  const float* conv1_b  = (const float*)d_in[14];
  const float* conv2_b  = (const float*)d_in[16];
  const float* ln1_g    = (const float*)d_in[17];
  const float* ln1_b    = (const float*)d_in[18];
  const float* ln2_g    = (const float*)d_in[19];
  const float* ln2_b    = (const float*)d_in[20];

  char* ws = (char*)d_ws;
  size_t off = 0;
  auto alloc = [&](size_t bytes) -> char* {
    char* p = ws + off;
    off += (bytes + 255) & ~(size_t)255;
    return p;
  };

  const size_t BSD = (size_t)BB * SS * DDIM;
  unsigned short* w_b16[8];   // transposed [N][K] = [512][512]
  for (int i = 0; i < 8; ++i) w_b16[i] = (unsigned short*)alloc((size_t)DDIM * DDIM * 2);
  unsigned short* c1_b16 = (unsigned short*)alloc((size_t)DDIM * DFF * 2);  // [2048][512]
  unsigned short* c2_b16 = (unsigned short*)alloc((size_t)DFF * DDIM * 2);  // [512][2048]
  unsigned short* xg  = (unsigned short*)alloc(BSD * 2);
  unsigned short* Qb  = (unsigned short*)alloc(BSD * 2);
  unsigned short* Kb  = (unsigned short*)alloc(BSD * 2);
  unsigned short* Vb  = (unsigned short*)alloc(BSD * 2);
  unsigned short* Ob  = (unsigned short*)alloc(BSD * 2);
  float* mha_f = (float*)alloc(BSD * 4);
  float* mha_r = (float*)alloc(BSD * 4);
  float* accum = (float*)alloc(BSD * 4);
  unsigned short* x1b = (unsigned short*)alloc(BSD * 2);
  unsigned short* mid = (unsigned short*)alloc((size_t)BB * SS * DFF * 2);
  float* x1 = mha_r;   // free after block-2 combine
  float* y2 = mha_f;   // free after block-2 combine

  // weights -> bf16 transposed
  for (int i = 0; i < 8; ++i) {
    int n = DDIM * DDIM;
    cvt_transpose_kernel<<<(n + 255) / 256, 256, 0, stream>>>(
        (const float*)d_in[5 + i], w_b16[i], DDIM, DDIM);
  }
  cvt_transpose_kernel<<<(DDIM * DFF + 255) / 256, 256, 0, stream>>>(
      (const float*)d_in[13], c1_b16, DDIM, DFF);   // conv1_w [D,DF] -> [DF][D]
  cvt_transpose_kernel<<<(DFF * DDIM + 255) / 256, 256, 0, stream>>>(
      (const float*)d_in[15], c2_b16, DFF, DDIM);   // conv2_w [DF,D] -> [D][DF]

  // accum = x
  hipMemcpyAsync(accum, x, BSD * 4, hipMemcpyDeviceToDevice, stream);

  const int M = BB * SS;
  const int tot4 = BB * SS * (DDIM / 4);
  dim3 gD(DDIM / 128, M / 128);     // N=512 GEMMs
  dim3 gA(SS / 64, HH, BB);

  auto run_mha = [&](const int* ie, int flip, unsigned short* wq, unsigned short* wk,
                     unsigned short* wv, unsigned short* wo, float* outp) {
    gather_bf16_kernel<<<(tot4 + 255) / 256, 256, 0, stream>>>(x, ie, xg, flip);
    gemm_bf16_kernel<4><<<gD, 128, 0, stream>>>(xg, wq, Qb, nullptr, M, DDIM, DDIM);  // Q pre-scaled
    gemm_bf16_kernel<1><<<gD, 128, 0, stream>>>(xg, wk, Kb, nullptr, M, DDIM, DDIM);
    gemm_bf16_kernel<1><<<gD, 128, 0, stream>>>(xg, wv, Vb, nullptr, M, DDIM, DDIM);
    attn_kernel<<<gA, 128, 0, stream>>>(Qb, Kb, Vb, Ob);
    gemm_bf16_kernel<0><<<gD, 128, 0, stream>>>(Ob, wo, outp, nullptr, M, DDIM, DDIM);
  };

  for (int blk = 0; blk < 2; ++blk) {
    run_mha(idx_emb[blk], 0, w_b16[0], w_b16[1], w_b16[2], w_b16[3], mha_f);
    run_mha(idx_emb[blk], 1, w_b16[4], w_b16[5], w_b16[6], w_b16[7], mha_r);
    addgather_kernel<<<(tot4 + 255) / 256, 256, 0, stream>>>(accum, mha_f, mha_r, idx_pro[blk]);
  }

  // x1 = LN(accum); also bf16 copy for FFN input
  ln_kernel<<<M, 256, 0, stream>>>(accum, nullptr, ln1_g, ln1_b, x1, x1b);
  // FFN
  gemm_bf16_kernel<2><<<dim3(DFF / 128, M / 128), 128, 0, stream>>>(x1b, c1_b16, mid, conv1_b, M, DFF, DDIM);
  gemm_bf16_kernel<3><<<dim3(DDIM / 128, M / 128), 128, 0, stream>>>(mid, c2_b16, y2, conv2_b, M, DDIM, DFF);
  // out = LN(x1 + y2)
  ln_kernel<<<M, 256, 0, stream>>>(x1, y2, ln2_g, ln2_b, (float*)d_out, nullptr);
}
